// SequenceModel_54296976556595
// MI455X (gfx1250) — matrix operand order repaired and checked by reading, not compiled, not run on hardware
//
#include <hip/hip_runtime.h>
#include <hip/hip_bf16.h>

typedef __attribute__((ext_vector_type(16))) _Float16 v16h;
typedef __attribute__((ext_vector_type(8)))  _Float16 v8h;
typedef __attribute__((ext_vector_type(8)))  float    v8f;

#define TSTEPS 1024
#define HDIM   64
#define HPAD   72   // padded f16 row stride: 144B -> conflict-free b128 LDS loads
#define OUTN   8

// gfx1250 V_TANH_F32 (confirmed lowering); fallback is division-free.
__device__ __forceinline__ float tanh_hw(float v) {
#if __has_builtin(__builtin_amdgcn_tanhf)
    return __builtin_amdgcn_tanhf(v);
#else
    float e = __builtin_amdgcn_exp2f(v * -2.885390082f);
    float r = __builtin_amdgcn_rcpf(1.0f + e);
    return __builtin_fmaf(2.0f, r, -1.0f);
#endif
}
// z already pre-scaled by 0.5 (folded into weights): sigmoid = 0.5*tanh(z/2)+0.5
__device__ __forceinline__ float sigm_prescaled(float zhalf) {
    return __builtin_fmaf(0.5f, tanh_hw(zhalf), 0.5f);
}

// B fragment: lane half holds K kb..kb+7 and kb+16..kb+23 of one Whh row,
// scaled (0.5 for sigmoid gates) and converted f32 -> f16.
__device__ __forceinline__ v16h load_b_frag(const float* __restrict__ W, int n, int kb, float sc) {
    const float4* p = (const float4*)(W + n * HDIM + kb);
    float4 q0 = p[0], q1 = p[1], q2 = p[4], q3 = p[5];
    v16h r;
    r[0]=(_Float16)(sc*q0.x); r[1]=(_Float16)(sc*q0.y); r[2]=(_Float16)(sc*q0.z); r[3]=(_Float16)(sc*q0.w);
    r[4]=(_Float16)(sc*q1.x); r[5]=(_Float16)(sc*q1.y); r[6]=(_Float16)(sc*q1.z); r[7]=(_Float16)(sc*q1.w);
    r[8]=(_Float16)(sc*q2.x); r[9]=(_Float16)(sc*q2.y); r[10]=(_Float16)(sc*q2.z); r[11]=(_Float16)(sc*q2.w);
    r[12]=(_Float16)(sc*q3.x); r[13]=(_Float16)(sc*q3.y); r[14]=(_Float16)(sc*q3.z); r[15]=(_Float16)(sc*q3.w);
    return r;
}

__global__ void __launch_bounds__(128)
lstm_seq2seq_kernel(const float* __restrict__ x,
                    const float* __restrict__ eWih, const float* __restrict__ eWhh,
                    const float* __restrict__ ebih, const float* __restrict__ ebhh,
                    const float* __restrict__ dWih, const float* __restrict__ dWhh,
                    const float* __restrict__ dbih, const float* __restrict__ dbhh,
                    const float* __restrict__ linW, const float* __restrict__ linb,
                    float* __restrict__ out)
{
    __shared__ __align__(16) _Float16 hbuf[2][16 * HPAD];
    __shared__ float sW[HDIM];
    __shared__ float sLb;

    const int tid  = threadIdx.x;
    const int wave = tid >> 5;
    const int lane = tid & 31;
    const int lo   = lane & 15;
    const bool hi  = lane >= 16;
    const int b0   = blockIdx.x * 16;
    const int j0   = wave * 16;

    for (int i = tid; i < 2 * 16 * HPAD; i += 128) (&hbuf[0][0])[i] = (_Float16)0.0f;
    if (tid < HDIM) sW[tid] = linW[tid];
    if (tid == 0)   sLb = linb[0];

    const int kb  = hi ? 8 : 0;    // B: same 16-bit 16x32 layout as A
    const int akb = hi ? 8  : 0;   // A: documented 16-bit 16x32 layout

    // gate scale: 0.5 for i/f/o (sigmoid via tanh), 1.0 for g (tanh)
    v16h Bf[4][2], B2[4];
#pragma unroll
    for (int g = 0; g < 4; ++g) {
        const float sc = (g == 2) ? 1.0f : 0.5f;
        const int n = g * HDIM + j0 + lo;
        Bf[g][0] = load_b_frag(eWhh, n, 0 + kb, sc);
        Bf[g][1] = load_b_frag(eWhh, n, 32 + kb, sc);
        v16h z2;
#pragma unroll
        for (int i = 0; i < 16; ++i) z2[i] = (_Float16)0.0f;
        if (!hi) {
            z2[0] = (_Float16)(sc * eWih[g * HDIM + j0 + lane]);   // K=64 row = Wih
            z2[1] = (_Float16)(sc * (ebih[n] + ebhh[n]));          // K=65 row = bias
        }
        B2[g] = z2;
    }

    v8f zc;                                   // zero accumulator seed (inline 0)
#pragma unroll
    for (int r = 0; r < 8; ++r) zc[r] = 0.0f;

    v16h ax;                                  // A chunk2: [x, 1.0, 0...] lanes<16
#pragma unroll
    for (int i = 0; i < 16; ++i) ax[i] = (_Float16)0.0f;
    if (!hi) ax[1] = (_Float16)1.0f;

    __syncthreads();

    float cst[8];
#pragma unroll
    for (int r = 0; r < 8; ++r) cst[r] = 0.0f;
    int cur = 0;

    // =================== encoder: 1024 steps ===================
    for (int t = 0; t < TSTEPS; ++t) {
        const _Float16* pa = &hbuf[cur][0] + lo * HPAD + akb;
        v8h a0l = *(const v8h*)(pa);
        v8h a0h = *(const v8h*)(pa + 16);
        v8h a1l = *(const v8h*)(pa + 32);
        v8h a1h = *(const v8h*)(pa + 48);
        v16h a0 = __builtin_shufflevector(a0l, a0h, 0,1,2,3,4,5,6,7,8,9,10,11,12,13,14,15);
        v16h a1 = __builtin_shufflevector(a1l, a1h, 0,1,2,3,4,5,6,7,8,9,10,11,12,13,14,15);

        if (!hi) ax[0] = (_Float16)x[(b0 + lane) * TSTEPS + t];

        v8f z[4];
#pragma unroll
        for (int g = 0; g < 4; ++g) {
            v8f acc;
            acc = __builtin_amdgcn_wmma_f32_16x16x32_f16(false, a0, false, Bf[g][0], (short)0, zc,  false, false);
            acc = __builtin_amdgcn_wmma_f32_16x16x32_f16(false, a1, false, Bf[g][1], (short)0, acc, false, false);
            acc = __builtin_amdgcn_wmma_f32_16x16x32_f16(false, ax, false, B2[g],   (short)0, acc, false, false);
            z[g] = acc;
        }

        const int nxt = cur ^ 1;
        _Float16* ho = &hbuf[nxt][(hi ? 8 : 0) * HPAD + j0 + lo];
#pragma unroll
        for (int r = 0; r < 8; ++r) {
            float iv = sigm_prescaled(z[0][r]);
            float fv = sigm_prescaled(z[1][r]);
            float gv = tanh_hw(z[2][r]);
            float ov = sigm_prescaled(z[3][r]);
            float cc = __builtin_fmaf(fv, cst[r], iv * gv);
            cst[r] = cc;
            ho[r * HPAD] = (_Float16)(ov * tanh_hw(cc));
        }
        __syncthreads();
        cur = nxt;
    }

    // =================== decoder setup ===================
    float xd = 0.0f;
    if (!hi) {   // xd = h_T @ linW + lin_b (redundant per wave; all waves need it)
        float s = sLb;
        const _Float16* hr = &hbuf[cur][lane * HPAD];
#pragma unroll
        for (int j = 0; j < HDIM; ++j) s += (float)hr[j] * sW[j];
        xd = s;
    }

#pragma unroll
    for (int g = 0; g < 4; ++g) {   // overwrite fragments with decoder weights
        const float sc = (g == 2) ? 1.0f : 0.5f;
        const int n = g * HDIM + j0 + lo;
        Bf[g][0] = load_b_frag(dWhh, n, 0 + kb, sc);
        Bf[g][1] = load_b_frag(dWhh, n, 32 + kb, sc);
        v16h z2;
#pragma unroll
        for (int i = 0; i < 16; ++i) z2[i] = (_Float16)0.0f;
        if (!hi) {
            z2[0] = (_Float16)(sc * dWih[g * HDIM + j0 + lane]);
            z2[1] = (_Float16)(sc * (dbih[n] + dbhh[n]));
        }
        B2[g] = z2;
    }

    // =================== decoder: 8 steps ===================
    for (int s = 0; s < OUTN; ++s) {
        const _Float16* pa = &hbuf[cur][0] + lo * HPAD + akb;
        v8h a0l = *(const v8h*)(pa);
        v8h a0h = *(const v8h*)(pa + 16);
        v8h a1l = *(const v8h*)(pa + 32);
        v8h a1h = *(const v8h*)(pa + 48);
        v16h a0 = __builtin_shufflevector(a0l, a0h, 0,1,2,3,4,5,6,7,8,9,10,11,12,13,14,15);
        v16h a1 = __builtin_shufflevector(a1l, a1h, 0,1,2,3,4,5,6,7,8,9,10,11,12,13,14,15);

        if (!hi) ax[0] = (_Float16)xd;

        v8f z[4];
#pragma unroll
        for (int g = 0; g < 4; ++g) {
            v8f acc;
            acc = __builtin_amdgcn_wmma_f32_16x16x32_f16(false, a0, false, Bf[g][0], (short)0, zc,  false, false);
            acc = __builtin_amdgcn_wmma_f32_16x16x32_f16(false, a1, false, Bf[g][1], (short)0, acc, false, false);
            acc = __builtin_amdgcn_wmma_f32_16x16x32_f16(false, ax, false, B2[g],   (short)0, acc, false, false);
            z[g] = acc;
        }

        const int nxt = cur ^ 1;
        _Float16* ho = &hbuf[nxt][(hi ? 8 : 0) * HPAD + j0 + lo];
#pragma unroll
        for (int r = 0; r < 8; ++r) {
            float iv = sigm_prescaled(z[0][r]);
            float fv = sigm_prescaled(z[1][r]);
            float gv = tanh_hw(z[2][r]);
            float ov = sigm_prescaled(z[3][r]);
            float cc = __builtin_fmaf(fv, cst[r], iv * gv);
            cst[r] = cc;
            ho[r * HPAD] = (_Float16)(ov * tanh_hw(cc));
        }
        __syncthreads();
        cur = nxt;

        // y = relu(h @ linW + lin_b); y feeds the next step
        float y = 0.0f;
        if (!hi) {
            float sum = sLb;
            const _Float16* hr = &hbuf[cur][lane * HPAD];
#pragma unroll
            for (int j = 0; j < HDIM; ++j) sum += (float)hr[j] * sW[j];
            y = fmaxf(sum, 0.0f);
            if (wave == 0) out[(b0 + lane) * OUTN + s] = y;
        }
        xd = y;
        __syncthreads();   // protect post-barrier read before next-step stores
    }
}

extern "C" void kernel_launch(void* const* d_in, const int* in_sizes, int n_in,
                              void* d_out, int out_size, void* d_ws, size_t ws_size,
                              hipStream_t stream) {
    (void)in_sizes; (void)n_in; (void)out_size; (void)d_ws; (void)ws_size;
    const float* x    = (const float*)d_in[0];
    const float* eWih = (const float*)d_in[1];
    const float* eWhh = (const float*)d_in[2];
    const float* ebih = (const float*)d_in[3];
    const float* ebhh = (const float*)d_in[4];
    const float* dWih = (const float*)d_in[5];
    const float* dWhh = (const float*)d_in[6];
    const float* dbih = (const float*)d_in[7];
    const float* dbhh = (const float*)d_in[8];
    const float* linW = (const float*)d_in[9];
    const float* linb = (const float*)d_in[10];
    float* out = (float*)d_out;

    const int B = 8192;
    dim3 grid(B / 16);   // 512 workgroups, 16 batch rows each
    dim3 block(128);     // 4 wave32s: one per 16-column gate slice
    lstm_seq2seq_kernel<<<grid, block, 0, stream>>>(
        x, eWih, eWhh, ebih, ebhh, dWih, dWhh, dbih, dbhh, linW, linb, out);
}
